// VRNN_11845519802805
// MI455X (gfx1250) — compile-verified, matrix operand order checked by reading, not run on hardware
//
#include <hip/hip_runtime.h>
#include <hip/hip_bf16.h>

// ---------------------------------------------------------------------------
// Types for CDNA5 WMMA (wave32, 16x16x32 bf16 -> f32)
// ---------------------------------------------------------------------------
typedef __bf16 bf16_t;
typedef __attribute__((ext_vector_type(16))) __bf16 v16bf;
typedef __attribute__((ext_vector_type(8)))  __bf16 v8bf;
typedef __attribute__((ext_vector_type(8)))  float  v8f;

union Frag16 { v16bf v; v8bf h[2]; };

// ---------------------------------------------------------------------------
// Problem constants (from the reference)
// ---------------------------------------------------------------------------
#define BB 32
#define TT 512
#define II 512
#define HH 1024
#define LL 2

// ---------------------------------------------------------------------------
// Generic WMMA GEMM:  C[M,N] = act( A[M,K] @ WT[N,K]^T + biasA + biasB + Dmat )
//   A, WT, Dmat bf16; fp32 accumulate; output bf16 or f32.
//   One 32x32 output tile per wave (2x2 grid of 16x16 WMMA tiles):
//     per k-step: 2 A-frags + 2 B-frags (8x global_load_b128) -> 4 WMMAs.
//   8 waves (256 threads) per block. M, N multiples of 32; K multiple of 32.
//
// CDNA5 operand layout (ISA 7.12.2, 16-bit A 16x32):
//   lane l (row = l&15, hi = l>>4): element e holds K = (e&8)*2 + hi*8 + (e&7)
//   -> elements 0..7 and 8..15 are each 8 contiguous bf16 (16B vector loads).
//   B mirrors A with N in place of M (WT stored row-major [N,K]).
//   C/D f32: col = tn*32 + 16*j + (l&15), row = tm*32 + 16*i + (l>>4)*8 + v (acc ij).
// ---------------------------------------------------------------------------
template <bool ADD_MAT, bool TANH_ACT, bool STORE_F32>
__global__ __launch_bounds__(256) void gemm_wmma_kernel(
    const bf16_t* __restrict__ A,     // [M,K] row-major bf16
    const bf16_t* __restrict__ WT,    // [N,K] row-major bf16 (W transposed)
    const float*  __restrict__ biasA, // [N] or nullptr
    const float*  __restrict__ biasB, // [N] or nullptr
    const bf16_t* __restrict__ Dmat,  // [M,N] bf16 pre-activation add, or nullptr
    void*         __restrict__ Cout,  // [M,N] bf16 or f32
    int M, int N, int K)
{
    const int lane = threadIdx.x & 31;
    const int wave = threadIdx.x >> 5;

    const int tilesN = N >> 5;                    // 32-wide N tiles
    const int totalTiles = (M >> 5) * tilesN;     // 32-tall M tiles
    const int gw = blockIdx.x * 8 + wave;
    if (gw >= totalTiles) return;                 // wave-uniform; EXEC stays all-1s

    const int tm = gw / tilesN;
    const int tn = gw - tm * tilesN;

    const int r  = lane & 15;
    const int hi = lane >> 4;

    const bf16_t* pa0 = A  + (size_t)(tm * 32 + r)      * K + hi * 8;
    const bf16_t* pa1 = A  + (size_t)(tm * 32 + 16 + r) * K + hi * 8;
    const bf16_t* pb0 = WT + (size_t)(tn * 32 + r)      * K + hi * 8;
    const bf16_t* pb1 = WT + (size_t)(tn * 32 + 16 + r) * K + hi * 8;

    v8f acc00 = {}, acc01 = {}, acc10 = {}, acc11 = {};

    const int ksteps = K >> 5;
    for (int kk = 0; kk < ksteps; ++kk) {
        Frag16 a0, a1, b0, b1;
        a0.h[0] = *(const v8bf*)(pa0);
        a0.h[1] = *(const v8bf*)(pa0 + 16);
        a1.h[0] = *(const v8bf*)(pa1);
        a1.h[1] = *(const v8bf*)(pa1 + 16);
        b0.h[0] = *(const v8bf*)(pb0);
        b0.h[1] = *(const v8bf*)(pb0 + 16);
        b1.h[0] = *(const v8bf*)(pb1);
        b1.h[1] = *(const v8bf*)(pb1 + 16);
        // Pull streaming operands ahead through global_prefetch_b8.
        __builtin_prefetch(pa0 + 256, 0, 1);
        __builtin_prefetch(pa1 + 256, 0, 1);

        acc00 = __builtin_amdgcn_wmma_f32_16x16x32_bf16(
            false, a0.v, false, b0.v, (short)0, acc00, false, false);
        acc01 = __builtin_amdgcn_wmma_f32_16x16x32_bf16(
            false, a0.v, false, b1.v, (short)0, acc01, false, false);
        acc10 = __builtin_amdgcn_wmma_f32_16x16x32_bf16(
            false, a1.v, false, b0.v, (short)0, acc10, false, false);
        acc11 = __builtin_amdgcn_wmma_f32_16x16x32_bf16(
            false, a1.v, false, b1.v, (short)0, acc11, false, false);

        pa0 += 32; pa1 += 32; pb0 += 32; pb1 += 32;
    }

    // ---- Epilogue (C/D VGPR layout) ----
    const int col0 = tn * 32 + r;
    const int col1 = col0 + 16;
    const int row0 = tm * 32 + hi * 8;       // M sub-tile 0
    const int row1 = row0 + 16;              // M sub-tile 1

    float bsum0 = 0.0f, bsum1 = 0.0f;
    if (biasA) { bsum0 += biasA[col0]; bsum1 += biasA[col1]; }
    if (biasB) { bsum0 += biasB[col0]; bsum1 += biasB[col1]; }

    const v8f accs[4] = { acc00, acc01, acc10, acc11 };
    const int rows[4] = { row0, row0, row1, row1 };
    const int cols[4] = { col0, col1, col0, col1 };
    const float bs[4] = { bsum0, bsum1, bsum0, bsum1 };

#pragma unroll
    for (int q = 0; q < 4; ++q) {
#pragma unroll
        for (int v = 0; v < 8; ++v) {
            const size_t row = (size_t)(rows[q] + v);
            const int    col = cols[q];
            float val = accs[q][v] + bs[q];
            if (ADD_MAT) val += (float)Dmat[row * N + col];
            if (TANH_ACT) val = tanhf(val);
            if (STORE_F32)
                ((float*)Cout)[row * N + col] = val;
            else
                ((bf16_t*)Cout)[row * N + col] = (bf16_t)val;
        }
    }
}

// ---------------------------------------------------------------------------
// Prep kernels: fp32 -> bf16 conversions / transposes (run once per launch)
// ---------------------------------------------------------------------------

// x[B,T,I] (f32) -> Xseq[(t*B+b), I] (bf16): batch-first -> time-major rows
__global__ void convert_x_kernel(const float* __restrict__ x,
                                 bf16_t* __restrict__ Xseq)
{
    size_t idx = (size_t)blockIdx.x * blockDim.x + threadIdx.x;
    const size_t total = (size_t)TT * BB * II;
    if (idx >= total) return;
    const int i  = (int)(idx % II);
    const size_t rt = idx / II;
    const int b = (int)(rt % BB);
    const int t = (int)(rt / BB);
    Xseq[idx] = (bf16_t)x[((size_t)b * TT + t) * II + i];
}

// W[K,N] (f32) -> WT[N,K] (bf16)
__global__ void transpose_w_kernel(const float* __restrict__ W,
                                   bf16_t* __restrict__ WT, int K, int N)
{
    size_t idx = (size_t)blockIdx.x * blockDim.x + threadIdx.x;
    const size_t total = (size_t)K * N;
    if (idx >= total) return;
    const int k = (int)(idx % K);
    const int n = (int)(idx / K);
    WT[idx] = (bf16_t)W[(size_t)k * N + n];
}

// plain f32 -> bf16 copy
__global__ void convert_f32_kernel(const float* __restrict__ src,
                                   bf16_t* __restrict__ dst, size_t n)
{
    size_t idx = (size_t)blockIdx.x * blockDim.x + threadIdx.x;
    if (idx >= n) return;
    dst[idx] = (bf16_t)src[idx];
}

// ---------------------------------------------------------------------------
// Host-side orchestration
// ---------------------------------------------------------------------------
static inline int gemm_blocks(int M, int N)
{
    const int tiles = (M / 32) * (N / 32);
    return (tiles + 7) / 8;
}

extern "C" void kernel_launch(void* const* d_in, const int* in_sizes, int n_in,
                              void* d_out, int out_size, void* d_ws, size_t ws_size,
                              hipStream_t stream)
{
    (void)in_sizes; (void)n_in; (void)out_size; (void)ws_size;

    const float* x    = (const float*)d_in[0];   // [B,T,I]
    const float* h0   = (const float*)d_in[1];   // [L,B,H]
    const float* Wi0  = (const float*)d_in[2];   // [I,H]
    const float* bi0  = (const float*)d_in[3];   // [H]
    const float* Wh0  = (const float*)d_in[4];   // [H,H]
    const float* bh0  = (const float*)d_in[5];   // [H]
    const float* Wi1  = (const float*)d_in[6];   // [H,H]
    const float* bi1  = (const float*)d_in[7];   // [H]
    const float* Wh1  = (const float*)d_in[8];   // [H,H]
    const float* bh1  = (const float*)d_in[9];   // [H]
    const float* Wlin = (const float*)d_in[10];  // [H,I]
    const float* blin = (const float*)d_in[11];  // [I]
    float* out = (float*)d_out;                  // [L,T,B,I] f32

    const int TB = TT * BB;  // 16384 rows

    // ---- workspace carve-up (all 256B aligned) ----
    char*  ws  = (char*)d_ws;
    size_t off = 0;
    auto carve = [&](size_t bytes) -> void* {
        void* p = ws + off;
        off = (off + bytes + 255) & ~(size_t)255;
        return p;
    };
    bf16_t* Xseq  = (bf16_t*)carve((size_t)TB * II * sizeof(bf16_t)); // 16 MB
    bf16_t* WiT0  = (bf16_t*)carve((size_t)HH * II * sizeof(bf16_t)); //  1 MB [H x I]
    bf16_t* WhT0  = (bf16_t*)carve((size_t)HH * HH * sizeof(bf16_t)); //  2 MB
    bf16_t* WiT1  = (bf16_t*)carve((size_t)HH * HH * sizeof(bf16_t)); //  2 MB
    bf16_t* WhT1  = (bf16_t*)carve((size_t)HH * HH * sizeof(bf16_t)); //  2 MB
    bf16_t* WlinT = (bf16_t*)carve((size_t)II * HH * sizeof(bf16_t)); //  1 MB [I x H]
    bf16_t* h0b   = (bf16_t*)carve((size_t)LL * BB * HH * sizeof(bf16_t));
    bf16_t* Xi0   = (bf16_t*)carve((size_t)TB * HH * sizeof(bf16_t)); // 32 MB
    bf16_t* out0  = (bf16_t*)carve((size_t)TB * HH * sizeof(bf16_t)); // 32 MB
    bf16_t* Xi1   = (bf16_t*)carve((size_t)TB * HH * sizeof(bf16_t)); // 32 MB
    bf16_t* out1  = (bf16_t*)carve((size_t)TB * HH * sizeof(bf16_t)); // 32 MB

    const int TPB = 256;
    auto blocks1d = [&](size_t n) { return (int)((n + TPB - 1) / TPB); };

    // ---- 1) convert / transpose everything to bf16 once ----
    convert_x_kernel<<<blocks1d((size_t)TB * II), TPB, 0, stream>>>(x, Xseq);
    transpose_w_kernel<<<blocks1d((size_t)II * HH), TPB, 0, stream>>>(Wi0,  WiT0,  II, HH);
    transpose_w_kernel<<<blocks1d((size_t)HH * HH), TPB, 0, stream>>>(Wh0,  WhT0,  HH, HH);
    transpose_w_kernel<<<blocks1d((size_t)HH * HH), TPB, 0, stream>>>(Wi1,  WiT1,  HH, HH);
    transpose_w_kernel<<<blocks1d((size_t)HH * HH), TPB, 0, stream>>>(Wh1,  WhT1,  HH, HH);
    transpose_w_kernel<<<blocks1d((size_t)HH * II), TPB, 0, stream>>>(Wlin, WlinT, HH, II);
    convert_f32_kernel<<<blocks1d((size_t)LL * BB * HH), TPB, 0, stream>>>(
        h0, h0b, (size_t)LL * BB * HH);

    // ---- 2) time-parallel precompute: Xi0 = Xseq @ Wi0 + bi0 + bh0 ----
    gemm_wmma_kernel<false, false, false><<<gemm_blocks(TB, HH), TPB, 0, stream>>>(
        Xseq, WiT0, bi0, bh0, nullptr, Xi0, TB, HH, II);

    // ---- 3) layer-0 recurrence: h_t = tanh(Xi0[t] + h_{t-1} @ Wh0) ----
    for (int t = 0; t < TT; ++t) {
        const bf16_t* hprev = (t == 0) ? h0b : out0 + (size_t)(t - 1) * BB * HH;
        gemm_wmma_kernel<true, true, false><<<gemm_blocks(BB, HH), TPB, 0, stream>>>(
            hprev, WhT0, nullptr, nullptr,
            Xi0 + (size_t)t * BB * HH, out0 + (size_t)t * BB * HH,
            BB, HH, HH);
    }

    // ---- 4) Xi1 = out0 @ Wi1 + bi1 + bh1 (time-parallel) ----
    gemm_wmma_kernel<false, false, false><<<gemm_blocks(TB, HH), TPB, 0, stream>>>(
        out0, WiT1, bi1, bh1, nullptr, Xi1, TB, HH, HH);

    // ---- 5) layer-1 recurrence ----
    for (int t = 0; t < TT; ++t) {
        const bf16_t* hprev = (t == 0) ? (h0b + (size_t)BB * HH)
                                       : out1 + (size_t)(t - 1) * BB * HH;
        gemm_wmma_kernel<true, true, false><<<gemm_blocks(BB, HH), TPB, 0, stream>>>(
            hprev, WhT1, nullptr, nullptr,
            Xi1 + (size_t)t * BB * HH, out1 + (size_t)t * BB * HH,
            BB, HH, HH);
    }

    // ---- 6) shared projection: y_l = out_l @ Wlin + blin (f32 out) ----
    gemm_wmma_kernel<false, false, true><<<gemm_blocks(TB, II), TPB, 0, stream>>>(
        out0, WlinT, blin, nullptr, nullptr, out, TB, II, HH);
    gemm_wmma_kernel<false, false, true><<<gemm_blocks(TB, II), TPB, 0, stream>>>(
        out1, WlinT, blin, nullptr, nullptr, out + (size_t)TB * II, TB, II, HH);
}
